// SamplerVQ_20641612824784
// MI455X (gfx1250) — matrix operand order repaired and checked by reading, not compiled, not run on hardware
//
#include <hip/hip_runtime.h>
#include <hip/hip_bf16.h>
#include <math.h>

typedef __bf16 v16bf __attribute__((ext_vector_type(16)));
typedef __bf16 v8bf  __attribute__((ext_vector_type(8)));
typedef __bf16 v4bf  __attribute__((ext_vector_type(4)));
typedef float  v8f   __attribute__((ext_vector_type(8)));
typedef unsigned int u32x4 __attribute__((ext_vector_type(4)));
typedef int          i32x8 __attribute__((ext_vector_type(8)));
typedef int          i32x4 __attribute__((ext_vector_type(4)));

#define DDIM 64
#define KCODES 512
#define NROWS (64 * 4096)          // 262144
#define TILE_M 256                 // rows per block iteration (8 waves x 32 rows)
#define NTILES (NROWS / TILE_M)    // 1024
#define ESTRIDE 72                 // bf16 codebook row stride (conflict-free)
#define XSTRF 68                   // fp32 x-row stride in dwords: 64 data + 4 TDM pad
#define NBLK 512
#define COMMIT 0.25f
#define EPSV 1e-10f
#define BIGC 256.0f                // positivity offset: ||x||^2 < 256 w.h.p. for chi^2_64

__device__ __forceinline__ v4bf f4_to_bf4(float4 v) {
    v4bf b;
    b[0] = (__bf16)v.x; b[1] = (__bf16)v.y; b[2] = (__bf16)v.z; b[3] = (__bf16)v.w;
    return b;
}

__device__ __forceinline__ v8bf cvt8(const float* p) {   // 8 fp32 -> 8 bf16
    float4 f0 = ((const float4*)p)[0];
    float4 f1 = ((const float4*)p)[1];
    v8bf b;
    b[0] = (__bf16)f0.x; b[1] = (__bf16)f0.y; b[2] = (__bf16)f0.z; b[3] = (__bf16)f0.w;
    b[4] = (__bf16)f1.x; b[5] = (__bf16)f1.y; b[6] = (__bf16)f1.z; b[7] = (__bf16)f1.w;
    return b;
}

__device__ __forceinline__ v16bf cat8(v8bf lo, v8bf hi) {
    return __builtin_shufflevector(lo, hi, 0,1,2,3,4,5,6,7,8,9,10,11,12,13,14,15);
}

__device__ __forceinline__ unsigned umin32(unsigned a, unsigned b) { return a < b ? a : b; }

__global__ void vq_zero_kernel(float* loss_acc, unsigned int* counts) {
    if (threadIdx.x == 0) *loss_acc = 0.0f;
    for (int i = threadIdx.x; i < KCODES; i += blockDim.x) counts[i] = 0u;
}

__global__ __launch_bounds__(256)
void vq_main_kernel(const float* __restrict__ x, const float* __restrict__ emb,
                    float* __restrict__ out, float* __restrict__ loss_acc,
                    unsigned int* __restrict__ counts) {
    __shared__ __bf16 ldse[KCODES * ESTRIDE];   // 73728 B (static)
    __shared__ float  enorm[KCODES];            //  2048 B (||e||^2 + BIGC)
    __shared__ int    rowidx[TILE_M];           //  1024 B
    __shared__ unsigned int hist[KCODES];       //  2048 B
    __shared__ float  lred[8];
    extern __shared__ float ldsxf[];            // dynamic: TILE_M x XSTRF fp32 (TDM target)

    const int tid  = threadIdx.x;
    const int lane = tid & 31;
    const int wave = tid >> 5;
    const int m    = lane & 15;
    const int half = lane >> 4;

    for (int i = tid; i < KCODES; i += 256) hist[i] = 0u;

    // ---- stage codebook -> bf16 LDS + (||e||^2 + BIGC), once per block ----
    for (int k = tid; k < KCODES; k += 256) {
        const float4* er = (const float4*)(emb + k * DDIM);
        float ss = 0.0f;
        #pragma unroll
        for (int q = 0; q < 16; ++q) {
            float4 v = er[q];
            ss += v.x*v.x + v.y*v.y + v.z*v.z + v.w*v.w;
            *(v4bf*)&ldse[k * ESTRIDE + q * 4] = f4_to_bf4(v);
        }
        enorm[k] = ss + BIGC;
    }
    __syncthreads();

    float lsum = 0.0f;

    for (int t = blockIdx.x; t < NTILES; t += gridDim.x) {
        const int rowBase = t * TILE_M;

        // ---- TDM: DMA 256x64 fp32 tile global->LDS with 4-DWORD row padding ----
        if (wave == 0) {
            unsigned long long ga =
                (unsigned long long)(const void*)(x + (size_t)rowBase * DDIM);
            unsigned ldsb = (unsigned)__builtin_amdgcn_groupstaticsize(); // dynamic LDS base
            u32x4 g0 = { 1u,                                   // count=1 valid descriptor
                         ldsb,                                 // lds_addr
                         (unsigned)ga,                         // global_addr[31:0]
                         (unsigned)((ga >> 32) & 0x1FFFFFFu) | (2u << 30) }; // addr[56:32]|type=2
            i32x8 g1 = { (int)((2u << 16)        // data_size = 4B
                             | (1u << 20)        // pad_enable
                             | (5u << 22)        // pad_interval: 64 DWORDs
                             | (3u << 25)),      // pad_amount:   4 DWORDs
                         (int)(64u << 16),       // tensor_dim0 = 64  (bits 79:48)
                         0,                      // dim0 hi / dim1 lo16
                         (int)((NROWS >> 16) | (64u << 16)), // dim1 hi16 | tile_dim0=64
                         256,                    // tile_dim1 = 256 rows
                         64,                     // tensor_dim0_stride = 64 elements
                         0, 0 };
            i32x4 gz4 = { 0, 0, 0, 0 };
            i32x8 gz8 = { 0, 0, 0, 0, 0, 0, 0, 0 };
            __builtin_amdgcn_tensor_load_to_lds(g0, g1, gz4, gz4, gz8, 0);
            __builtin_amdgcn_s_wait_tensorcnt(0);
        }
        __syncthreads();

        // ---- wave owns 32 rows: build bf16 A tiles from fp32 LDS ----
        const int ab = half * 8;
        const float* xr0 = &ldsxf[(wave * 32 + m) * XSTRF];
        const float* xr1 = &ldsxf[(wave * 32 + 16 + m) * XSTRF];
        v16bf a00 = cat8(cvt8(xr0 + ab),      cvt8(xr0 + ab + 16));
        v16bf a01 = cat8(cvt8(xr0 + 32 + ab), cvt8(xr0 + 32 + ab + 16));
        v16bf a10 = cat8(cvt8(xr1 + ab),      cvt8(xr1 + ab + 16));
        v16bf a11 = cat8(cvt8(xr1 + 32 + ab), cvt8(xr1 + 32 + ab + 16));

        unsigned minu0[8], minu1[8];
        #pragma unroll
        for (int j = 0; j < 8; ++j) { minu0[j] = 0xFFFFFFFFu; minu1[j] = 0xFFFFFFFFu; }

        #pragma unroll 2
        for (int tn = 0; tn < KCODES / 16; ++tn) {
            const int ncode = tn * 16 + m;
            const __bf16* brow = &ldse[ncode * ESTRIDE];
            const int bb = half * 8;
            v16bf b0 = cat8(*(const v8bf*)&brow[bb],      *(const v8bf*)&brow[bb + 16]);
            v16bf b1 = cat8(*(const v8bf*)&brow[32 + bb], *(const v8bf*)&brow[32 + bb + 16]);

            v8f acc0 = {}, acc1 = {};
            acc0 = __builtin_amdgcn_wmma_f32_16x16x32_bf16(false, a00, false, b0,
                                                           (short)0, acc0, false, false);
            acc1 = __builtin_amdgcn_wmma_f32_16x16x32_bf16(false, a10, false, b0,
                                                           (short)0, acc1, false, false);
            acc0 = __builtin_amdgcn_wmma_f32_16x16x32_bf16(false, a01, false, b1,
                                                           (short)0, acc0, false, false);
            acc1 = __builtin_amdgcn_wmma_f32_16x16x32_bf16(false, a11, false, b1,
                                                           (short)0, acc1, false, false);

            const float enC = enorm[ncode];   // ||e||^2 + BIGC (scores stay >= 0)
            const unsigned code = (unsigned)ncode;
            #pragma unroll
            for (int j = 0; j < 8; ++j) {
                float s0 = fmaf(-2.0f, acc0[j], enC);
                float s1 = fmaf(-2.0f, acc1[j], enC);
                unsigned u0 = (__float_as_uint(s0) & 0xFFFFFE00u) | code;
                unsigned u1 = (__float_as_uint(s1) & 0xFFFFFE00u) | code;
                minu0[j] = umin32(minu0[j], u0);
                minu1[j] = umin32(minu1[j], u1);
            }
        }

        // ---- cross-lane argmin (pure umin; index rides in the low 9 bits) ----
        #pragma unroll
        for (int j = 0; j < 8; ++j) {
            #pragma unroll
            for (int mask = 8; mask >= 1; mask >>= 1) {
                minu0[j] = umin32(minu0[j], (unsigned)__shfl_xor((int)minu0[j], mask, 16));
                minu1[j] = umin32(minu1[j], (unsigned)__shfl_xor((int)minu1[j], mask, 16));
            }
        }
        if (m == 0) {   // lanes 0 (rows j / j+16) and 16 (rows j+8 / j+24)
            #pragma unroll
            for (int j = 0; j < 8; ++j) {
                rowidx[wave * 32 + half * 8 + j]      = (int)(minu0[j] & 0x1FFu);
                rowidx[wave * 32 + 16 + half * 8 + j] = (int)(minu1[j] & 0x1FFu);
            }
        }
        __syncthreads();

        // ---- gather + straight-through output + loss (x read from fp32 LDS) ----
        for (int c = tid; c < TILE_M * (DDIM / 4); c += 256) {
            int r = c >> 4, q = c & 15;
            int code = rowidx[r];
            float4 xv = ((const float4*)&ldsxf[r * XSTRF])[q];
            float4 ev = ((const float4*)(emb + (size_t)code * DDIM))[q];
            float dx = ev.x - xv.x, dy = ev.y - xv.y, dz = ev.z - xv.z, dw = ev.w - xv.w;
            float4 o; o.x = xv.x + dx; o.y = xv.y + dy; o.z = xv.z + dz; o.w = xv.w + dw;
            ((float4*)(out + (size_t)(rowBase + r) * DDIM))[q] = o;
            lsum += dx*dx + dy*dy + dz*dz + dw*dw;
        }
        atomicAdd(&hist[rowidx[tid]], 1u);   // one row per thread (TILE_M == blockDim)
        __syncthreads();                      // protects ldsxf/rowidx for next TDM/tile
    }

    // ---- block reductions -> global accumulators ----
    #pragma unroll
    for (int mask = 16; mask >= 1; mask >>= 1) lsum += __shfl_xor(lsum, mask, 32);
    if (lane == 0) lred[wave] = lsum;
    __syncthreads();
    if (wave == 0) {
        float v = (lane < 8) ? lred[lane] : 0.0f;
        #pragma unroll
        for (int mask = 4; mask >= 1; mask >>= 1) v += __shfl_xor(v, mask, 8);
        if (lane == 0) atomicAdd(loss_acc, v);
    }
    for (int i = tid; i < KCODES; i += 256)
        if (hist[i]) atomicAdd(&counts[i], hist[i]);
}

__global__ void vq_finalize_kernel(const float* __restrict__ loss_acc,
                                   const unsigned int* __restrict__ counts,
                                   float* __restrict__ out_tail) {
    __shared__ float red[16];
    const int tid = threadIdx.x;          // 512 threads, one per code
    float p = (float)counts[tid] / (float)NROWS;
    float term = -p * logf(p + EPSV);
    #pragma unroll
    for (int mask = 16; mask >= 1; mask >>= 1) term += __shfl_xor(term, mask, 32);
    if ((tid & 31) == 0) red[tid >> 5] = term;
    __syncthreads();
    if (tid == 0) {
        float s = 0.0f;
        #pragma unroll
        for (int i = 0; i < 16; ++i) s += red[i];
        out_tail[0] = COMMIT * (*loss_acc) / (float)((size_t)NROWS * DDIM);  // loss
        out_tail[1] = expf(s);                                               // perplexity
    }
}

extern "C" void kernel_launch(void* const* d_in, const int* in_sizes, int n_in,
                              void* d_out, int out_size, void* d_ws, size_t ws_size,
                              hipStream_t stream) {
    const float* x   = (const float*)d_in[0];   // [64,4096,64] fp32
    const float* emb = (const float*)d_in[1];   // [512,64]     fp32
    float* out = (float*)d_out;                 // q_st flat + [loss, perplexity]

    float* loss_acc      = (float*)d_ws;
    unsigned int* counts = (unsigned int*)d_ws + 1;

    const size_t xTileBytes = (size_t)TILE_M * XSTRF * sizeof(float);  // 69632 B dynamic LDS

    vq_zero_kernel<<<1, 512, 0, stream>>>(loss_acc, counts);
    vq_main_kernel<<<NBLK, 256, xTileBytes, stream>>>(x, emb, out, loss_acc, counts);
    vq_finalize_kernel<<<1, KCODES, 0, stream>>>(loss_acc, counts,
                                                 out + (size_t)NROWS * DDIM);
}
